// GraphTransformerNet_429496729670
// MI455X (gfx1250) — matrix operand order repaired and checked by reading, not compiled, not run on hardware
//
#include <hip/hip_runtime.h>
#include <math.h>

#define FIN  128
#define D1C  256
#define D2C  128
#define HEADS_C 4

typedef _Float16 half4v  __attribute__((ext_vector_type(4)));
typedef _Float16 half8v  __attribute__((ext_vector_type(8)));
typedef _Float16 half16v __attribute__((ext_vector_type(16)));
typedef float    float8v __attribute__((ext_vector_type(8)));

#define LDS_STRIDE 40   // halves; 80 B rows -> 16B-aligned fragments, conflict-friendly

// ---------------------------------------------------------------------------
// WMMA GEMM:  C[M,Nc] = A[M,K] (f32 row-major) @ B[K,Nc] (f32 row-major) + bias
// 256 threads = 8 waves. Block tile 128x64; each wave: 16 rows x 64 cols via
// FOUR v_wmma_f32_16x16x32_f16 per 32-wide K-step (A fragment reused 4x).
// A staged [row][k], B staged transposed [col][k] in LDS as f16 so each lane
// fragment is 2x ds_load_b128. Next K-tile prefetched into registers
// (branchless: row clamped + cndmask zero) to overlap global latency w/ WMMA.
// ---------------------------------------------------------------------------
__global__ __launch_bounds__(256) void wmma_gemm_bias(
    const float* __restrict__ A, const float* __restrict__ B,
    const float* __restrict__ bias, float* __restrict__ C,
    int M, int K, int Nc)
{
  __shared__ __align__(16) _Float16 As[128 * LDS_STRIDE];  // 10.0 KB
  __shared__ __align__(16) _Float16 Bs[64 * LDS_STRIDE];   //  5.0 KB
  const int tid  = threadIdx.x;
  const int wave = tid >> 5;
  const int lane = tid & 31;
  const int rowBase = blockIdx.x * 128;
  const int colBase = blockIdx.y * 64;

  // staging assignment: thread covers (row = p*32 + tr, k = tc4..tc4+3)
  const int tr  = tid >> 3;        // 0..31
  const int tc4 = (tid & 7) * 4;   // 0,4,...,28

  float4 aReg[4];
  float4 bReg[2];

  auto loadTile = [&](int k0) {
#pragma unroll
    for (int p = 0; p < 4; ++p) {
      int gr = rowBase + p * 32 + tr;
      bool ok = gr < M;
      int grc = ok ? gr : (M - 1);               // clamped: always issue the load
      float4 v = *(const float4*)(A + (size_t)grc * K + k0 + tc4);
      if (!ok) v = float4{0.0f, 0.0f, 0.0f, 0.0f};   // v_cndmask, no exec branch
      aReg[p] = v;
    }
    const float* brow = B + (size_t)(k0 + tr) * Nc + colBase + tc4;
    bReg[0] = *(const float4*)(brow);
    bReg[1] = *(const float4*)(brow + 32);
  };
  auto stageTile = [&]() {
#pragma unroll
    for (int p = 0; p < 4; ++p) {
      int r = p * 32 + tr;
      half4v h = { (_Float16)aReg[p].x, (_Float16)aReg[p].y,
                   (_Float16)aReg[p].z, (_Float16)aReg[p].w };
      *(half4v*)(&As[r * LDS_STRIDE + tc4]) = h;   // 8B ds_store
    }
    // B transposed: Bs[col][kk]
#pragma unroll
    for (int hh = 0; hh < 2; ++hh) {
      _Float16 hb[4] = { (_Float16)bReg[hh].x, (_Float16)bReg[hh].y,
                         (_Float16)bReg[hh].z, (_Float16)bReg[hh].w };
#pragma unroll
      for (int j = 0; j < 4; ++j)
        Bs[(hh * 32 + tc4 + j) * LDS_STRIDE + tr] = hb[j];
    }
  };

  float8v acc[4] = {{}, {}, {}, {}};

  // fragment geometry (ISA 7.12.2):
  // A: lane row = wave*16 + lane%16; K = {0..7,16..23} + 8*(lane>=16)
  // B: lane col = lane%16 (+16t for tile t); K = {0..7,16..23} + 8*(lane>=16)
  const int rloc = (wave << 4) + (lane & 15);
  const int kahi = (lane >> 4) << 3;
  const int bcol = lane & 15;
  const int kbhi = (lane >> 4) << 3;

  loadTile(0);
  for (int k0 = 0; k0 < K; k0 += 32) {
    stageTile();
    __syncthreads();
    if (k0 + 32 < K) loadTile(k0 + 32);   // prefetch next tile (overlaps WMMA)

    const _Float16* arow = &As[rloc * LDS_STRIDE + kahi];
    half8v alo = *(const half8v*)(arow);        // K = kahi..kahi+7
    half8v ahi = *(const half8v*)(arow + 16);   // K = kahi+16..kahi+23
    half16v a = __builtin_shufflevector(alo, ahi, 0,1,2,3,4,5,6,7,8,9,10,11,12,13,14,15);

    half16v bfrag[4];
#pragma unroll
    for (int t = 0; t < 4; ++t) {
      const _Float16* bp = &Bs[(bcol + 16 * t) * LDS_STRIDE + kbhi];
      half8v lo = *(const half8v*)(bp);
      half8v hi = *(const half8v*)(bp + 16);
      bfrag[t] = __builtin_shufflevector(lo, hi, 0,1,2,3,4,5,6,7,8,9,10,11,12,13,14,15);
    }
#pragma unroll
    for (int t = 0; t < 4; ++t)
      acc[t] = __builtin_amdgcn_wmma_f32_16x16x32_f16(false, a, false, bfrag[t], (short)0, acc[t], false, false);
    __syncthreads();
  }

  // C/D layout: VGPR r holds M = r + 8*(lane>=16), N = lane%16 (+16t)
  const int ocol = lane & 15;
  const int rsub = (lane >> 4) << 3;
#pragma unroll
  for (int r = 0; r < 8; ++r) {
    int gr = rowBase + (wave << 4) + r + rsub;
    if (gr < M) {
      float* crow = C + (size_t)gr * Nc + colBase + ocol;
      const float* brow = bias + colBase + ocol;
#pragma unroll
      for (int t = 0; t < 4; ++t)
        crow[16 * t] = acc[t][r] + brow[16 * t];
    }
  }
}

// ---------------------------------------------------------------------------
// helpers
// ---------------------------------------------------------------------------
__device__ inline void atomicMaxF(float* addr, float val) {
  unsigned int* ua = (unsigned int*)addr;
  unsigned int old = *ua;
  while (__uint_as_float(old) < val) {
    unsigned int assumed = old;
    old = atomicCAS(ua, assumed, __float_as_uint(val));
    if (old == assumed) break;
  }
}

__global__ void fill_kernel(float* __restrict__ p, float v, int n) {
  int i = blockIdx.x * blockDim.x + threadIdx.x;
  if (i < n) p[i] = v;
}

__device__ inline float dot4(const float* __restrict__ a, const float* __restrict__ b, int n) {
  float acc = 0.0f;
  for (int i = 0; i < n; i += 4) {
    float4 av = *(const float4*)(a + i);
    float4 bv = *(const float4*)(b + i);
    acc += av.x * bv.x + av.y * bv.y + av.z * bv.z + av.w * bv.w;
  }
  return acc;
}

// ---------------------------------------------------------------------------
// edge attention kernels
// ---------------------------------------------------------------------------
template <int H, int DH>
__global__ void edge_score_max(const int* __restrict__ src, const int* __restrict__ dst,
                               const float* __restrict__ q, const float* __restrict__ k,
                               float* __restrict__ score, float* __restrict__ mmax,
                               int E, float scale)
{
  int t = blockIdx.x * blockDim.x + threadIdx.x;
  if (t >= E * H) return;
  int e = t / H, h = t - e * H;
  int s = src[e], d = dst[e];
  const float* qp = q + (size_t)d * (H * DH) + h * DH;
  const float* kp = k + (size_t)s * (H * DH) + h * DH;
  float acc;
  if (DH % 4 == 0) {
    acc = dot4(qp, kp, DH);
  } else {
    acc = 0.0f;
#pragma unroll
    for (int i = 0; i < DH; ++i) acc += qp[i] * kp[i];
  }
  acc *= scale;
  score[t] = acc;
  atomicMaxF(&mmax[(size_t)d * H + h], acc);
}

template <int H>
__global__ void edge_exp_sum(const int* __restrict__ dst, float* __restrict__ score,
                             const float* __restrict__ mmax, float* __restrict__ denom, int E)
{
  int t = blockIdx.x * blockDim.x + threadIdx.x;
  if (t >= E * H) return;
  int e = t / H, h = t - e * H;
  int d = dst[e];
  float p = __expf(score[t] - mmax[(size_t)d * H + h]);
  score[t] = p;
  atomicAdd(&denom[(size_t)d * H + h], p);
}

template <int H, int DH>
__global__ void edge_aggregate(const int* __restrict__ src, const int* __restrict__ dst,
                               const float* __restrict__ score, const float* __restrict__ denom,
                               const float* __restrict__ v, float* __restrict__ outAgg, int E)
{
  int t = blockIdx.x * blockDim.x + threadIdx.x;
  if (t >= E * H) return;
  int e = t / H, h = t - e * H;
  int s = src[e], d = dst[e];
  float alpha = score[t] / (denom[(size_t)d * H + h] + 1e-16f);
  const float* vp = v + (size_t)s * (H * DH) + h * DH;
  float* op = outAgg + (size_t)d * (H * DH) + h * DH;
#pragma unroll
  for (int i = 0; i < DH; ++i) atomicAdd(&op[i], alpha * vp[i]);
}

// ---------------------------------------------------------------------------
// beta gating + BatchNorm(eval) + ELU, one wave32 per node
// ---------------------------------------------------------------------------
template <int D>
__global__ __launch_bounds__(256) void gate_bn_elu(
    const float* __restrict__ agg, const float* __restrict__ xr,
    const float* __restrict__ Wb,  const float* __restrict__ g,
    const float* __restrict__ be,  const float* __restrict__ mean,
    const float* __restrict__ var, float* __restrict__ y, int n)
{
  const int wave = threadIdx.x >> 5, lane = threadIdx.x & 31;
  const int node = blockIdx.x * 8 + wave;
  if (node >= n) return;
  constexpr int C = D / 32;
  float o[C], x[C];
  float part = 0.0f;
#pragma unroll
  for (int c = 0; c < C; ++c) {
    int d = lane + (c << 5);
    float ov = agg[(size_t)node * D + d];
    float xv = xr[(size_t)node * D + d];
    o[c] = ov; x[c] = xv;
    part += Wb[d] * ov + Wb[D + d] * xv + Wb[2 * D + d] * (ov - xv);
  }
#pragma unroll
  for (int off = 16; off > 0; off >>= 1) part += __shfl_xor(part, off, 32);
  float beta = 1.0f / (1.0f + __expf(-part));
#pragma unroll
  for (int c = 0; c < C; ++c) {
    int d = lane + (c << 5);
    float yv = beta * x[c] + (1.0f - beta) * o[c];
    yv = (yv - mean[d]) * rsqrtf(var[d] + 1e-5f) * g[d] + be[d];
    yv = yv > 0.0f ? yv : (__expf(yv) - 1.0f);
    y[(size_t)node * D + d] = yv;
  }
}

// ---------------------------------------------------------------------------
// reconstruction loss: per-edge 128-dim dot + BCE term, atomic accumulate
// ---------------------------------------------------------------------------
__global__ void loss_kernel(const float* __restrict__ z, const int* __restrict__ si,
                            const int* __restrict__ di, float* __restrict__ acc,
                            int E, int isNeg)
{
  int e = blockIdx.x * blockDim.x + threadIdx.x;
  if (e >= E) return;
  const float* a = z + (size_t)si[e] * D2C;
  const float* b = z + (size_t)di[e] * D2C;
  float dot = dot4(a, b, D2C);
  float s = 1.0f / (1.0f + __expf(-dot));
  float term = isNeg ? -__logf(1.0f - s + 1e-15f) : -__logf(s + 1e-15f);
  atomicAdd(&acc[isNeg], term);
}

// ---------------------------------------------------------------------------
// conv3 (heads=1, dh=1): four 128->1 GEMVs per node
// ---------------------------------------------------------------------------
__global__ void conv3_proj(const float* __restrict__ z,
                           const float* __restrict__ Wq, const float* __restrict__ bq,
                           const float* __restrict__ Wk, const float* __restrict__ bk,
                           const float* __restrict__ Wv, const float* __restrict__ bv,
                           const float* __restrict__ Ws, const float* __restrict__ bs,
                           float* __restrict__ q3, float* __restrict__ k3,
                           float* __restrict__ v3, float* __restrict__ s3, int n)
{
  int i = blockIdx.x * blockDim.x + threadIdx.x;
  if (i >= n) return;
  const float* zp = z + (size_t)i * D2C;
  float aq = 0, ak = 0, av = 0, as_ = 0;
#pragma unroll 4
  for (int d = 0; d < D2C; d += 4) {
    float4 zv = *(const float4*)(zp + d);
    float4 wq = *(const float4*)(Wq + d);
    float4 wk = *(const float4*)(Wk + d);
    float4 wv = *(const float4*)(Wv + d);
    float4 ws = *(const float4*)(Ws + d);
    aq  += zv.x * wq.x + zv.y * wq.y + zv.z * wq.z + zv.w * wq.w;
    ak  += zv.x * wk.x + zv.y * wk.y + zv.z * wk.z + zv.w * wk.w;
    av  += zv.x * wv.x + zv.y * wv.y + zv.z * wv.z + zv.w * wv.w;
    as_ += zv.x * ws.x + zv.y * ws.y + zv.z * ws.z + zv.w * ws.w;
  }
  q3[i] = aq + bq[0]; k3[i] = ak + bk[0]; v3[i] = av + bv[0]; s3[i] = as_ + bs[0];
}

__global__ void gate3_kernel(const float* __restrict__ out3, const float* __restrict__ s3,
                             const float* __restrict__ Wb3, float* __restrict__ xout, int n)
{
  int i = blockIdx.x * blockDim.x + threadIdx.x;
  if (i >= n) return;
  float o = out3[i], x = s3[i];
  float beta = 1.0f / (1.0f + __expf(-(Wb3[0] * o + Wb3[1] * x + Wb3[2] * (o - x))));
  xout[i] = beta * x + (1.0f - beta) * o;
}

__global__ void finalize_kernel(const float* __restrict__ acc,
                                const float* __restrict__ c1, const float* __restrict__ c2,
                                float* __restrict__ out, int n, float invE)
{
  if (threadIdx.x == 0 && blockIdx.x == 0) {
    out[n]     = acc[0] * invE + acc[1] * invE;
    out[n + 1] = c1[0];
    out[n + 2] = c2[0];
  }
}

// ---------------------------------------------------------------------------
// launch
// ---------------------------------------------------------------------------
static inline int cdiv(int a, int b) { return (a + b - 1) / b; }

extern "C" void kernel_launch(void* const* d_in, const int* in_sizes, int n_in,
                              void* d_out, int out_size, void* d_ws, size_t ws_size,
                              hipStream_t stream)
{
  const float* x    = (const float*)d_in[0];
  const int*   ei   = (const int*)d_in[1];
  const int*   nei  = (const int*)d_in[2];
  const float *Wq1 = (const float*)d_in[3],  *bq1 = (const float*)d_in[4];
  const float *Wk1 = (const float*)d_in[5],  *bk1 = (const float*)d_in[6];
  const float *Wv1 = (const float*)d_in[7],  *bv1 = (const float*)d_in[8];
  const float *Ws1 = (const float*)d_in[9],  *bs1 = (const float*)d_in[10];
  const float *Wb1 = (const float*)d_in[11];
  const float *g1  = (const float*)d_in[12], *be1 = (const float*)d_in[13];
  const float *m1  = (const float*)d_in[14], *v1  = (const float*)d_in[15];
  const float *Wq2 = (const float*)d_in[16], *bq2 = (const float*)d_in[17];
  const float *Wk2 = (const float*)d_in[18], *bk2 = (const float*)d_in[19];
  const float *Wv2 = (const float*)d_in[20], *bv2 = (const float*)d_in[21];
  const float *Ws2 = (const float*)d_in[22], *bs2 = (const float*)d_in[23];
  const float *Wb2 = (const float*)d_in[24];
  const float *g2  = (const float*)d_in[25], *be2 = (const float*)d_in[26];
  const float *m2  = (const float*)d_in[27], *v2  = (const float*)d_in[28];
  const float *Wq3 = (const float*)d_in[29], *bq3 = (const float*)d_in[30];
  const float *Wk3 = (const float*)d_in[31], *bk3 = (const float*)d_in[32];
  const float *Wv3 = (const float*)d_in[33], *bv3 = (const float*)d_in[34];
  const float *Ws3 = (const float*)d_in[35], *bs3 = (const float*)d_in[36];
  const float *Wb3 = (const float*)d_in[37];
  const float *c1  = (const float*)d_in[38], *c2 = (const float*)d_in[39];

  const int N = in_sizes[0] / FIN;      // 50000
  const int E = in_sizes[1] / 2;        // 250000
  const int* src = ei;       const int* dst = ei + E;
  const int* ns  = nei;      const int* nd  = nei + E;

  float* ws = (float*)d_ws;
  size_t off = 0;
  auto alloc = [&](size_t n) { float* p = ws + off; off += n; return p; };
  float* Pq   = alloc((size_t)N * D1C);   // q1 -> reused as q2
  float* Pk   = alloc((size_t)N * D1C);   // k1 -> k2
  float* Pv   = alloc((size_t)N * D1C);   // v1 -> v2
  float* Ps   = alloc((size_t)N * D1C);   // s1 -> s2
  float* Agg  = alloc((size_t)N * D1C);   // out1 -> out2
  float* Hbuf = alloc((size_t)N * D1C);   // h -> z
  float* Mx   = alloc((size_t)N * HEADS_C);
  float* Dn   = alloc((size_t)N * HEADS_C);
  float* Sc   = alloc((size_t)E * HEADS_C);
  float* q3   = alloc(N); float* k3 = alloc(N); float* v3 = alloc(N); float* s3 = alloc(N);
  float* out3 = alloc(N); float* m3 = alloc(N); float* d3 = alloc(N);
  float* sc3  = alloc(E);
  float* accL = alloc(2);

  const dim3 blk(256);
  const float NEG_INF = -3.0e38f;

  // ------------------ conv1: projections (WMMA GEMMs) ------------------
  {
    dim3 grid(cdiv(N, 128), D1C / 64);
    wmma_gemm_bias<<<grid, blk, 0, stream>>>(x, Wq1, bq1, Pq, N, FIN, D1C);
    wmma_gemm_bias<<<grid, blk, 0, stream>>>(x, Wk1, bk1, Pk, N, FIN, D1C);
    wmma_gemm_bias<<<grid, blk, 0, stream>>>(x, Wv1, bv1, Pv, N, FIN, D1C);
    wmma_gemm_bias<<<grid, blk, 0, stream>>>(x, Ws1, bs1, Ps, N, FIN, D1C);
  }
  // ------------------ conv1: edge softmax aggregation ------------------
  {
    int nh = N * HEADS_C, eh = E * HEADS_C, nd1 = N * D1C;
    fill_kernel<<<cdiv(nh, 256), blk, 0, stream>>>(Mx, NEG_INF, nh);
    fill_kernel<<<cdiv(nh, 256), blk, 0, stream>>>(Dn, 0.0f, nh);
    fill_kernel<<<cdiv(nd1, 256), blk, 0, stream>>>(Agg, 0.0f, nd1);
    float scale = 1.0f / sqrtf(64.0f);
    edge_score_max<HEADS_C, 64><<<cdiv(eh, 256), blk, 0, stream>>>(src, dst, Pq, Pk, Sc, Mx, E, scale);
    edge_exp_sum<HEADS_C><<<cdiv(eh, 256), blk, 0, stream>>>(dst, Sc, Mx, Dn, E);
    edge_aggregate<HEADS_C, 64><<<cdiv(eh, 256), blk, 0, stream>>>(src, dst, Sc, Dn, Pv, Agg, E);
    gate_bn_elu<D1C><<<cdiv(N, 8), blk, 0, stream>>>(Agg, Ps, Wb1, g1, be1, m1, v1, Hbuf, N);
  }
  // ------------------ conv2: projections (WMMA GEMMs) ------------------
  {
    dim3 grid(cdiv(N, 128), D2C / 64);
    wmma_gemm_bias<<<grid, blk, 0, stream>>>(Hbuf, Wq2, bq2, Pq, N, D1C, D2C);
    wmma_gemm_bias<<<grid, blk, 0, stream>>>(Hbuf, Wk2, bk2, Pk, N, D1C, D2C);
    wmma_gemm_bias<<<grid, blk, 0, stream>>>(Hbuf, Wv2, bv2, Pv, N, D1C, D2C);
    wmma_gemm_bias<<<grid, blk, 0, stream>>>(Hbuf, Ws2, bs2, Ps, N, D1C, D2C);
  }
  // ------------------ conv2: edge softmax aggregation -> z ------------------
  {
    int nh = N * HEADS_C, eh = E * HEADS_C, nd2 = N * D2C;
    fill_kernel<<<cdiv(nh, 256), blk, 0, stream>>>(Mx, NEG_INF, nh);
    fill_kernel<<<cdiv(nh, 256), blk, 0, stream>>>(Dn, 0.0f, nh);
    fill_kernel<<<cdiv(nd2, 256), blk, 0, stream>>>(Agg, 0.0f, nd2);
    float scale = 1.0f / sqrtf(32.0f);
    edge_score_max<HEADS_C, 32><<<cdiv(eh, 256), blk, 0, stream>>>(src, dst, Pq, Pk, Sc, Mx, E, scale);
    edge_exp_sum<HEADS_C><<<cdiv(eh, 256), blk, 0, stream>>>(dst, Sc, Mx, Dn, E);
    edge_aggregate<HEADS_C, 32><<<cdiv(eh, 256), blk, 0, stream>>>(src, dst, Sc, Dn, Pv, Agg, E);
    gate_bn_elu<D2C><<<cdiv(N, 8), blk, 0, stream>>>(Agg, Ps, Wb2, g2, be2, m2, v2, Hbuf, N);
  }
  // ------------------ reconstruction loss ------------------
  {
    fill_kernel<<<1, blk, 0, stream>>>(accL, 0.0f, 2);
    loss_kernel<<<cdiv(E, 256), blk, 0, stream>>>(Hbuf, src, dst, accL, E, 0);
    loss_kernel<<<cdiv(E, 256), blk, 0, stream>>>(Hbuf, ns, nd, accL, E, 1);
  }
  // ------------------ conv3 (heads=1, dh=1) ------------------
  {
    conv3_proj<<<cdiv(N, 256), blk, 0, stream>>>(Hbuf, Wq3, bq3, Wk3, bk3, Wv3, bv3,
                                                 Ws3, bs3, q3, k3, v3, s3, N);
    fill_kernel<<<cdiv(N, 256), blk, 0, stream>>>(m3, NEG_INF, N);
    fill_kernel<<<cdiv(N, 256), blk, 0, stream>>>(d3, 0.0f, N);
    fill_kernel<<<cdiv(N, 256), blk, 0, stream>>>(out3, 0.0f, N);
    edge_score_max<1, 1><<<cdiv(E, 256), blk, 0, stream>>>(src, dst, q3, k3, sc3, m3, E, 1.0f);
    edge_exp_sum<1><<<cdiv(E, 256), blk, 0, stream>>>(dst, sc3, m3, d3, E);
    edge_aggregate<1, 1><<<cdiv(E, 256), blk, 0, stream>>>(src, dst, sc3, d3, v3, out3, E);
    gate3_kernel<<<cdiv(N, 256), blk, 0, stream>>>(out3, s3, Wb3, (float*)d_out, N);
  }
  // ------------------ outputs: r_loss, c1, c2 ------------------
  finalize_kernel<<<1, 64, 0, stream>>>(accL, c1, c2, (float*)d_out, N, 1.0f / (float)E);
}
